// SupervisedSimCSE_3865470566962
// MI455X (gfx1250) — matrix operand order repaired and checked by reading, not compiled, not run on hardware
//
#include <hip/hip_runtime.h>
#include <hip/hip_bf16.h>

typedef _Float16 v16h __attribute__((ext_vector_type(16)));
typedef _Float16 v8h  __attribute__((ext_vector_type(8)));
typedef _Float16 v4h  __attribute__((ext_vector_type(4)));
typedef float    v8f  __attribute__((ext_vector_type(8)));
typedef int      v4i  __attribute__((ext_vector_type(4)));

union F16x16 { v16h v; v8h h[2]; };

#define DIMD    1024
#define NROW    4096
#define NCAND   8192
#define APITCH  1032          // halves: 1024 + 8 pad -> 516-dword stride, conflict-free
#define KC      128           // B K-chunk (halves)
#define NKC     (DIMD / KC)   // 8 chunks
#define BPITCH  (KC + 8)      // 136 halves = 68 dwords stride
#define BBUF    (64 * BPITCH) // halves per B buffer
#define MBLK    128           // rows per workgroup (4 M-waves x 32)
#define NSPLIT  8             // 128 N-blocks of 64 cols / 8 splits = 16 per WG
#define NSLICE  (NSPLIT * 2)  // sumexp partial slices (per wn half)

#define AS1 __attribute__((address_space(1)))
#define AS3 __attribute__((address_space(3)))

#if defined(__has_builtin)
#if __has_builtin(__builtin_amdgcn_global_load_async_to_lds_b128)
#define HAVE_ASYNC_LDS 1
#endif
#endif
#ifndef HAVE_ASYNC_LDS
#define HAVE_ASYNC_LDS 0
#endif

// ---------------------------------------------------------------------------
// Kernel 1: normalize rows (cosine eps clamp) and convert to f16.
// ---------------------------------------------------------------------------
__global__ void simcse_normalize(const float* __restrict__ s,
                                 const float* __restrict__ p,
                                 const float* __restrict__ n,
                                 _Float16* __restrict__ sh,
                                 _Float16* __restrict__ candh) {
    const int row = blockIdx.x;             // 0..12287
    const float* src;
    _Float16* dst;
    if (row < NROW)            { src = s + (size_t)row * DIMD;            dst = sh    + (size_t)row * DIMD; }
    else if (row < 2 * NROW)   { src = p + (size_t)(row - NROW) * DIMD;   dst = candh + (size_t)(row - NROW) * DIMD; }
    else                       { src = n + (size_t)(row - 2*NROW) * DIMD; dst = candh + (size_t)(row - NROW) * DIMD; }

    const int t = threadIdx.x;              // 256 threads, 4 floats each
    float4 v = ((const float4*)src)[t];
    float ss = v.x * v.x + v.y * v.y + v.z * v.z + v.w * v.w;
    #pragma unroll
    for (int i = 16; i >= 1; i >>= 1) ss += __shfl_xor(ss, i, 32);

    __shared__ float wsum[8];
    if ((t & 31) == 0) wsum[t >> 5] = ss;
    __syncthreads();
    float tot = 0.f;
    #pragma unroll
    for (int i = 0; i < 8; ++i) tot += wsum[i];

    const float scale = 1.f / fmaxf(sqrtf(tot), 1e-8f);
    v4h h;
    h.x = (_Float16)(v.x * scale);
    h.y = (_Float16)(v.y * scale);
    h.z = (_Float16)(v.z * scale);
    h.w = (_Float16)(v.w * scale);
    ((v4h*)dst)[t] = h;
}

// ---------------------------------------------------------------------------
// async / fallback B-chunk staging: 64 cand rows x KC halves -> LDS buffer
// ---------------------------------------------------------------------------
__device__ __forceinline__ void stage_b_chunk(const _Float16* __restrict__ gsrc,
                                              _Float16* ldst, int tid) {
    #pragma unroll
    for (int it = 0; it < 4; ++it) {
        const int idx = tid + 256 * it;        // 1024 x 16B chunks
        const int r = idx >> 4;                // row 0..63
        const int c = (idx & 15) << 3;         // half offset 0..120
        const _Float16* g = gsrc + (size_t)r * DIMD + c;
        _Float16* l = ldst + r * BPITCH + c;
#if HAVE_ASYNC_LDS
        __builtin_amdgcn_global_load_async_to_lds_b128(
            (AS1 v4i*)g, (AS3 v4i*)l, 0, 0);
#else
        *(v8h*)l = *(const v8h*)g;
#endif
    }
}

__device__ __forceinline__ void async_wait_le4() {
#if HAVE_ASYNC_LDS
#if __has_builtin(__builtin_amdgcn_s_wait_asynccnt)
    __builtin_amdgcn_s_wait_asynccnt(4);
#else
    asm volatile("s_wait_asynccnt 0x4" ::: "memory");
#endif
#endif
}
__device__ __forceinline__ void async_wait_le0() {
#if HAVE_ASYNC_LDS
#if __has_builtin(__builtin_amdgcn_s_wait_asynccnt)
    __builtin_amdgcn_s_wait_asynccnt(0);
#else
    asm volatile("s_wait_asynccnt 0x0" ::: "memory");
#endif
#endif
}

// ---------------------------------------------------------------------------
// Kernel 2: fused f16 WMMA GEMM (2x2 register-blocked) + sumexp epilogue.
// grid = (32 M-blocks, 8 N-splits), block = 256 (8 waves; wave = (wm, wn)).
// Wave computes a 32x32 output block per 64-col N-block; B double-buffered
// in K-chunks via async LDS copies; A resident full-K in LDS.
// ---------------------------------------------------------------------------
extern __shared__ _Float16 smem[];

__global__ void __launch_bounds__(256)
simcse_gemm_sumexp(const _Float16* __restrict__ sh,
                   const _Float16* __restrict__ candh,
                   float* __restrict__ Spart,     // [NSLICE][NROW]
                   float* __restrict__ diag) {    // [NROW]
    const int mb   = blockIdx.x;        // 0..31
    const int ns   = blockIdx.y;        // 0..7
    const int tid  = threadIdx.x;
    const int wave = tid >> 5;
    const int lane = tid & 31;
    const int wm   = wave >> 1;         // 0..3 : M sub-block of 32 rows
    const int wn   = wave & 1;          // 0..1 : N half of 32 cols

    _Float16* Alds = smem;                        // MBLK x APITCH
    _Float16* Blds = smem + MBLK * APITCH;        // 2 x (64 x BPITCH)

    const int rowBase = mb * MBLK;

    // ---- stage A (128 rows x 1024 halves), coalesced 16B chunks ----
    for (int i = tid; i < MBLK * (DIMD / 8); i += 256) {
        const int r = i >> 7;
        const int c = (i & 127) << 3;
        *(v8h*)&Alds[r * APITCH + c] = *(const v8h*)&sh[(size_t)(rowBase + r) * DIMD + c];
    }
    __syncthreads();

    float sacc[2][8];
    #pragma unroll
    for (int i = 0; i < 2; ++i)
        #pragma unroll
        for (int r = 0; r < 8; ++r) sacc[i][r] = 0.f;
    float dv0 = 0.f, dv1 = 0.f;

    const bool owns_diag = (ns == (mb >> 3)) && (wn == (wm & 1));
    const int  diag_nb   = 2 * mb + (wm >> 1);

    const int asel = (lane < 16) ? 0 : 8;    // ISA 16-bit A-layout lane K shift
    const int bsel = (lane < 16) ? 0 : 8;    // ISA 16-bit B-layout lane K shift
    const _Float16* Arow0 = &Alds[(32 * wm + (lane & 15)) * APITCH];
    const _Float16* Arow1 = Arow0 + 16 * APITCH;
    const _Float16* Bcol  = &Blds[(32 * wn + (lane & 15)) * BPITCH];

    const int nb0 = ns * 16, nb1 = nb0 + 16;
    #pragma unroll 1
    for (int nb = nb0; nb < nb1; ++nb) {
        const _Float16* gB = candh + (size_t)(64 * nb) * DIMD;

        v8f acc00 = 0.f, acc01 = 0.f, acc10 = 0.f, acc11 = 0.f;

        stage_b_chunk(gB, Blds, tid);                       // chunk 0 -> buf0
        for (int kc = 0; kc < NKC; ++kc) {
            if (kc + 1 < NKC) {
                stage_b_chunk(gB + (kc + 1) * KC, Blds + ((kc + 1) & 1) * BBUF, tid);
                async_wait_le4();   // oldest chunk (kc) complete; newest may fly
            } else {
                async_wait_le0();
            }
            __syncthreads();        // all waves see chunk kc in LDS

            const _Float16* Brow0 = Bcol + (kc & 1) * BBUF;
            const _Float16* Brow1 = Brow0 + 16 * BPITCH;
            const int gk = kc * KC;
            #pragma unroll
            for (int k0 = 0; k0 < KC; k0 += 32) {
                F16x16 a0, a1, b0, b1;
                a0.h[0] = *(const v8h*)&Arow0[gk + k0 + asel];
                a0.h[1] = *(const v8h*)&Arow0[gk + k0 + asel + 16];
                a1.h[0] = *(const v8h*)&Arow1[gk + k0 + asel];
                a1.h[1] = *(const v8h*)&Arow1[gk + k0 + asel + 16];
                b0.h[0] = *(const v8h*)&Brow0[k0 + bsel];
                b0.h[1] = *(const v8h*)&Brow0[k0 + bsel + 16];
                b1.h[0] = *(const v8h*)&Brow1[k0 + bsel];
                b1.h[1] = *(const v8h*)&Brow1[k0 + bsel + 16];
                acc00 = __builtin_amdgcn_wmma_f32_16x16x32_f16(false, a0.v, false, b0.v, (short)0, acc00, false, false);
                acc01 = __builtin_amdgcn_wmma_f32_16x16x32_f16(false, a0.v, false, b1.v, (short)0, acc01, false, false);
                acc10 = __builtin_amdgcn_wmma_f32_16x16x32_f16(false, a1.v, false, b0.v, (short)0, acc10, false, false);
                acc11 = __builtin_amdgcn_wmma_f32_16x16x32_f16(false, a1.v, false, b1.v, (short)0, acc11, false, false);
            }
            __syncthreads();        // done reading buf[kc&1] before it is restaged
        }

        // diagonal dots live in sub-tiles (0,0) and (1,1) of the matching block
        if (owns_diag && nb == diag_nb) {
            #pragma unroll
            for (int r = 0; r < 8; ++r) {
                if (lane == r)      { dv0 = acc00[r]; dv1 = acc11[r]; }  // rows 0..7 of tile
                if (lane == 24 + r) { dv0 = acc00[r]; dv1 = acc11[r]; }  // rows 8..15
            }
        }

        // epilogue: logits = 20*dot <= 20 -> fixed-max sumexp
        #pragma unroll
        for (int r = 0; r < 8; ++r) {
            sacc[0][r] += __expf(acc00[r] * 20.f - 20.f) + __expf(acc01[r] * 20.f - 20.f);
            sacc[1][r] += __expf(acc10[r] * 20.f - 20.f) + __expf(acc11[r] * 20.f - 20.f);
        }
    }

    // reduce columns across the 16 lanes sharing each row half
    #pragma unroll
    for (int i = 0; i < 2; ++i)
        #pragma unroll
        for (int r = 0; r < 8; ++r) {
            #pragma unroll
            for (int d = 1; d < 16; d <<= 1)
                sacc[i][r] += __shfl_xor(sacc[i][r], d, 32);
        }

    const int gRowW  = rowBase + 32 * wm;
    const int pslice = ns * 2 + wn;
    if (lane == 0) {
        #pragma unroll
        for (int i = 0; i < 2; ++i)
            #pragma unroll
            for (int r = 0; r < 8; ++r)
                Spart[(size_t)pslice * NROW + gRowW + 16 * i + r] = sacc[i][r];
    }
    if (lane == 16) {
        #pragma unroll
        for (int i = 0; i < 2; ++i)
            #pragma unroll
            for (int r = 0; r < 8; ++r)
                Spart[(size_t)pslice * NROW + gRowW + 16 * i + 8 + r] = sacc[i][r];
    }
    if (owns_diag) {
        if (lane < 8) {
            diag[gRowW + lane]      = dv0;
            diag[gRowW + 16 + lane] = dv1;
        } else if (lane >= 24) {
            diag[gRowW + 8 + (lane - 24)]  = dv0;
            diag[gRowW + 24 + (lane - 24)] = dv1;
        }
    }
}

// ---------------------------------------------------------------------------
// Kernel 3: combine partials -> mean loss (single block, deterministic).
// ---------------------------------------------------------------------------
__global__ void simcse_finalize(const float* __restrict__ Spart,
                                const float* __restrict__ diag,
                                float* __restrict__ out) {
    const int tid = threadIdx.x;  // 256
    float local = 0.f;
    for (int i = tid; i < NROW; i += 256) {
        float ssum = 0.f;
        #pragma unroll
        for (int p = 0; p < NSLICE; ++p) ssum += Spart[(size_t)p * NROW + i];
        local += 20.f + __logf(ssum) - 20.f * diag[i];
    }
    __shared__ float red[256];
    red[tid] = local;
    __syncthreads();
    #pragma unroll
    for (int off = 128; off >= 1; off >>= 1) {
        if (tid < off) red[tid] += red[tid + off];
        __syncthreads();
    }
    if (tid == 0) out[0] = red[0] * (1.f / (float)NROW);
}

// ---------------------------------------------------------------------------
extern "C" void kernel_launch(void* const* d_in, const int* in_sizes, int n_in,
                              void* d_out, int out_size, void* d_ws, size_t ws_size,
                              hipStream_t stream) {
    const float* s = (const float*)d_in[0];
    const float* p = (const float*)d_in[1];
    const float* n = (const float*)d_in[2];

    _Float16* sh    = (_Float16*)d_ws;                        //  8 MB
    _Float16* candh = sh + (size_t)NROW * DIMD;               // 16 MB
    float*    Spart = (float*)(candh + (size_t)NCAND * DIMD); // 256 KB
    float*    diag  = Spart + (size_t)NSLICE * NROW;          //  16 KB

    simcse_normalize<<<3 * NROW, 256, 0, stream>>>(s, p, n, sh, candh);

    const size_t lds = (size_t)(MBLK * APITCH + 2 * BBUF) * sizeof(_Float16); // ~299 KB
    simcse_gemm_sumexp<<<dim3(NROW / MBLK, NSPLIT), 256, lds, stream>>>(sh, candh, Spart, diag);

    simcse_finalize<<<1, 256, 0, stream>>>(Spart, diag, (float*)d_out);
}
